// ErnieLayoutEncoder_9251359556007
// MI455X (gfx1250) — matrix operand order repaired and checked by reading, not compiled, not run on hardware
//
#include <hip/hip_runtime.h>
#include <hip/hip_bf16.h>

#define BB 2
#define SS 2048
#define DD 768
#define HH 12
#define DHH 64
#define LLAYERS 12

typedef __bf16 bf16x16 __attribute__((ext_vector_type(16)));
typedef float f32x8 __attribute__((ext_vector_type(8)));

union Frag {
    bf16x16 v;
    unsigned short u16[16];
    uint4 q[2];
};

static __device__ inline unsigned short f2bf(float f) {
    union { float f; unsigned int u; } x; x.f = f;
    unsigned int r = x.u + 0x7FFFu + ((x.u >> 16) & 1u);
    return (unsigned short)(r >> 16);
}

static __device__ inline f32x8 wmma_bf16(const Frag& a, const Frag& b, f32x8 c) {
    return __builtin_amdgcn_wmma_f32_16x16x32_bf16(false, a.v, false, b.v,
                                                   (short)0, c, false, false);
}

// ---------------- prep kernels ----------------

// wt layout: [4][L][D][D] bf16, wt[w][l][n][k] = W[w][l][k][n]
__global__ void k_prep_w(const float* __restrict__ qw, const float* __restrict__ kw,
                         const float* __restrict__ vw, const float* __restrict__ ow,
                         unsigned short* __restrict__ wt) {
    long long idx = (long long)blockIdx.x * 256 + threadIdx.x;
    const long long total = 4LL * LLAYERS * DD * DD;
    if (idx >= total) return;
    int k = (int)(idx % DD); long long t = idx / DD;
    int n = (int)(t % DD); t /= DD;
    int l = (int)(t % LLAYERS); int w = (int)(t / LLAYERS);
    const float* src = (w == 0) ? qw : (w == 1) ? kw : (w == 2) ? vw : ow;
    wt[idx] = f2bf(src[((long long)l * DD + k) * DD + n]);
}

// T[h][p][x][y] = rp[h][p] + rx[h][x] + ry[h][y]  (h*131072 floats)
__global__ void k_table(const float* __restrict__ rp, const float* __restrict__ rx,
                        const float* __restrict__ ry, float* __restrict__ T) {
    int idx = blockIdx.x * 256 + threadIdx.x;
    if (idx >= HH * 32 * 64 * 64) return;
    int y = idx & 63, x = (idx >> 6) & 63, p = (idx >> 12) & 31, h = idx >> 17;
    T[idx] = rp[h * 32 + p] + rx[h * 64 + x] + ry[h * 64 + y];
}

static __device__ inline int rel_bucket(int rel, int nbuckets, float maxdist) {
    int nb = nbuckets >> 1;
    int ret = rel > 0 ? nb : 0;
    int n = rel < 0 ? -rel : rel;
    int max_exact = nb >> 1;
    if (n < max_exact) return ret + n;
    int nn = n < 1 ? 1 : n;
    float v = __logf((float)nn / (float)max_exact) /
              __logf(maxdist / (float)max_exact) * (float)(nb - max_exact);
    int vi = max_exact + (int)v;
    if (vi > nb - 1) vi = nb - 1;
    return ret + vi;
}

__global__ void k_pidx(const int* __restrict__ pos, const int* __restrict__ bbox,
                       unsigned int* __restrict__ pidx) {
    long long idx = (long long)blockIdx.x * 256 + threadIdx.x;
    if (idx >= (long long)BB * SS * SS) return;
    int k = (int)(idx % SS); long long t = idx / SS;
    int q = (int)(t % SS); int b = (int)(t / SS);
    int rp = pos[b * SS + k] - pos[b * SS + q];                       // memory - query
    int rx = bbox[(b * SS + k) * 4 + 0] - bbox[(b * SS + q) * 4 + 0];
    int ry = bbox[(b * SS + k) * 4 + 3] - bbox[(b * SS + q) * 4 + 3];
    int bp = rel_bucket(rp, 32, 128.f);
    int bx = rel_bucket(rx, 64, 256.f);
    int by = rel_bucket(ry, 64, 256.f);
    pidx[idx] = (unsigned)(bp * 4096 + bx * 64 + by);
}

__global__ void k_init_h(const float* __restrict__ hs, float* __restrict__ hf,
                         unsigned short* __restrict__ hb) {
    int idx = blockIdx.x * 256 + threadIdx.x;
    if (idx >= BB * SS * DD) return;
    float v = hs[idx];
    hf[idx] = v;
    hb[idx] = f2bf(v);
}

// ---------------- QKV projection GEMM ----------------
// One wave computes a 32x64 strip of one of Q/K/V; double-buffered fragments.
// waves = 3 * (4096/32) * (768/64) = 4608 -> 576 blocks.
__global__ __launch_bounds__(256) void k_gemm_qkv(
    const unsigned short* __restrict__ hb,    // [B*S, D] bf16
    const unsigned short* __restrict__ wt,    // [4][L][D][D] bf16 transposed
    const float* __restrict__ qb, const float* __restrict__ kb,
    const float* __restrict__ vb,
    unsigned short* __restrict__ qo,          // [B,H,S,DH]
    unsigned short* __restrict__ ko,          // [B,H,S,DH]
    unsigned short* __restrict__ vto,         // [B,H,DH,S]
    int layer) {
    int wave = (int)((blockIdx.x * blockDim.x + threadIdx.x) >> 5);
    int lane = threadIdx.x & 31;
    int lh = lane >> 4, ln = lane & 15;
    const int WPM = (BB * SS / 32) * (DD / 64);   // 1536 waves per matrix
    int mat = wave / WPM;
    int rem = wave % WPM;
    int mtile = rem / (DD / 64);
    int ntile = rem % (DD / 64);

    const unsigned short* W = wt + ((long long)mat * LLAYERS + layer) * DD * DD;
    const float* bias = ((mat == 0) ? qb : (mat == 1) ? kb : vb) + layer * DD;

    f32x8 acc[2][4];
    for (int i = 0; i < 2; ++i)
        for (int j = 0; j < 4; ++j)
            for (int r = 0; r < 8; ++r) acc[i][j][r] = 0.f;

    const unsigned short* arow0 = hb + (long long)(mtile * 32 + ln) * DD + lh * 8;
    const unsigned short* arow1 = arow0 + 16LL * DD;
    const unsigned short* bptr[4];
#pragma unroll
    for (int nt = 0; nt < 4; ++nt)
        bptr[nt] = W + (long long)(ntile * 64 + nt * 16 + ln) * DD + lh * 8;

    auto load_a = [&](Frag* a, int k0) {
        a[0].q[0] = *(const uint4*)(arow0 + k0);
        a[0].q[1] = *(const uint4*)(arow0 + k0 + 16);
        a[1].q[0] = *(const uint4*)(arow1 + k0);
        a[1].q[1] = *(const uint4*)(arow1 + k0 + 16);
    };
    auto load_b = [&](Frag* b, int k0) {
#pragma unroll
        for (int nt = 0; nt < 4; ++nt) {
            b[nt].q[0] = *(const uint4*)(bptr[nt] + k0);
            b[nt].q[1] = *(const uint4*)(bptr[nt] + k0 + 16);
        }
    };
    auto mmas = [&](const Frag* a, const Frag* b) {
#pragma unroll
        for (int mi = 0; mi < 2; ++mi)
#pragma unroll
            for (int nt = 0; nt < 4; ++nt)
                acc[mi][nt] = wmma_bf16(a[mi], b[nt], acc[mi][nt]);
    };

    Frag aA[2], bA[4], aB[2], bB[4];
    load_a(aA, 0);
    load_b(bA, 0);
    for (int kk = 0; kk < DD / 32; kk += 2) {
        load_a(aB, (kk + 1) * 32);
        load_b(bB, (kk + 1) * 32);
        mmas(aA, bA);
        if (kk + 2 < DD / 32) {
            load_a(aA, (kk + 2) * 32);
            load_b(bA, (kk + 2) * 32);
        }
        mmas(aB, bB);
    }

    const float scale = (mat == 0) ? 0.125f : 1.0f;   // 1/sqrt(64) on Q
#pragma unroll
    for (int mi = 0; mi < 2; ++mi)
#pragma unroll
        for (int nt = 0; nt < 4; ++nt)
#pragma unroll
            for (int r = 0; r < 8; ++r) {
                int m = mi * 16 + r + 8 * lh;
                int row = mtile * 32 + m;
                int b = row / SS, s = row % SS;
                int n = ntile * 64 + nt * 16 + ln;
                float val = (acc[mi][nt][r] + bias[n]) * scale;
                int h = n / DHH, dh = n % DHH;
                unsigned short bv = f2bf(val);
                if (mat == 0)      qo[(((long long)b * HH + h) * SS + s) * DHH + dh] = bv;
                else if (mat == 1) ko[(((long long)b * HH + h) * SS + s) * DHH + dh] = bv;
                else               vto[(((long long)b * HH + h) * DHH + dh) * SS + s] = bv;
            }
}

// ---------------- flash attention ----------------
// One wave per (b, h, 16-query tile). 3072 waves = 384 blocks x 8 waves.
// K/V fragments double-buffered across 32-key blocks.
__global__ __launch_bounds__(256) void k_attn(
    const unsigned short* __restrict__ q, const unsigned short* __restrict__ k,
    const unsigned short* __restrict__ vt, const unsigned int* __restrict__ pidx,
    const float* __restrict__ T, const unsigned char* __restrict__ mask,
    unsigned short* __restrict__ ctx) {
    __shared__ __align__(16) unsigned short lds[8][2][16][16];  // per-wave P staging
    int wib = threadIdx.x >> 5;
    int wave = (int)((blockIdx.x * blockDim.x + threadIdx.x) >> 5);
    int lane = threadIdx.x & 31;
    int lh = lane >> 4, ln = lane & 15;

    int qtile = wave % (SS / 16);
    int bh = wave / (SS / 16);
    int h = bh % HH;
    int b = bh / HH;
    int q0 = qtile * 16;

    const unsigned short* qbase = q + ((long long)bh * SS + q0) * DHH;
    const unsigned short* kbase = k + (long long)bh * SS * DHH;
    const unsigned short* vbase = vt + (long long)bh * DHH * SS;
    const float* Th = T + h * (32 * 64 * 64);
    const unsigned int* pxb = pidx + ((long long)b * SS + q0) * SS;
    const unsigned char* mkb = mask + b * SS;

    // preload Q A-fragments for contraction K = 0..31 and 32..63
    Frag aq0, aq1;
    {
        const unsigned short* qr = qbase + (long long)ln * DHH;
        aq0.q[0] = *(const uint4*)(qr + lh * 8);
        aq0.q[1] = *(const uint4*)(qr + lh * 8 + 16);
        aq1.q[0] = *(const uint4*)(qr + 32 + lh * 8);
        aq1.q[1] = *(const uint4*)(qr + 32 + lh * 8 + 16);
    }

    float rm[8], rl[8];
    f32x8 oacc[4];
    for (int r = 0; r < 8; ++r) { rm[r] = -3.0e38f; rl[r] = 0.f; }
    for (int nt = 0; nt < 4; ++nt) for (int r = 0; r < 8; ++r) oacc[nt][r] = 0.f;

    // bk layout: [half*2 + kstep]
    auto load_k = [&](Frag* bk, int kbas) {
#pragma unroll
        for (int half = 0; half < 2; ++half) {
            const unsigned short* kr =
                kbase + (long long)(kbas + half * 16 + ln) * DHH + lh * 8;
            bk[half * 2 + 0].q[0] = *(const uint4*)(kr);
            bk[half * 2 + 0].q[1] = *(const uint4*)(kr + 16);
            bk[half * 2 + 1].q[0] = *(const uint4*)(kr + 32);
            bk[half * 2 + 1].q[1] = *(const uint4*)(kr + 48);
        }
    };
    auto load_v = [&](Frag* bv, int kbas) {
#pragma unroll
        for (int nt = 0; nt < 4; ++nt) {
            const unsigned short* vr =
                vbase + (long long)(nt * 16 + ln) * SS + kbas + lh * 8;
            bv[nt].q[0] = *(const uint4*)vr;
            bv[nt].q[1] = *(const uint4*)(vr + 16);
        }
    };

    auto attn_block = [&](const Frag* bk, const Frag* bv, int kbas) {
        f32x8 c[2];
#pragma unroll
        for (int half = 0; half < 2; ++half) {
            f32x8 cc;
            for (int r = 0; r < 8; ++r) cc[r] = 0.f;
            cc = wmma_bf16(aq0, bk[half * 2 + 0], cc);
            cc = wmma_bf16(aq1, bk[half * 2 + 1], cc);
            int kidx = kbas + half * 16 + ln;
            bool msk = mkb[kidx] != 0;
#pragma unroll
            for (int r = 0; r < 8; ++r) {
                int m = r + 8 * lh;
                float bvl = Th[pxb[(long long)m * SS + kidx]];
                cc[r] = msk ? -3.0e38f : (cc[r] + bvl);
            }
            c[half] = cc;
        }
        // online softmax (row reductions live in each 16-lane half)
        float sf[8];
#pragma unroll
        for (int r = 0; r < 8; ++r) {
            float tmax = fmaxf(c[0][r], c[1][r]);
            tmax = fmaxf(tmax, __shfl_xor(tmax, 1, 32));
            tmax = fmaxf(tmax, __shfl_xor(tmax, 2, 32));
            tmax = fmaxf(tmax, __shfl_xor(tmax, 4, 32));
            tmax = fmaxf(tmax, __shfl_xor(tmax, 8, 32));
            float nm = fmaxf(rm[r], tmax);
            float sc = __expf(rm[r] - nm);
            float p0 = __expf(c[0][r] - nm);
            float p1 = __expf(c[1][r] - nm);
            float rs = p0 + p1;
            rs += __shfl_xor(rs, 1, 32);
            rs += __shfl_xor(rs, 2, 32);
            rs += __shfl_xor(rs, 4, 32);
            rs += __shfl_xor(rs, 8, 32);
            rl[r] = rl[r] * sc + rs;
            rm[r] = nm;
            c[0][r] = p0; c[1][r] = p1;
            sf[r] = sc;
        }
#pragma unroll
        for (int nt = 0; nt < 4; ++nt)
#pragma unroll
            for (int r = 0; r < 8; ++r) oacc[nt][r] *= sf[r];

        // C-layout -> A-layout transpose of P through LDS (bf16)
        __syncthreads();
#pragma unroll
        for (int r = 0; r < 8; ++r) {
            int m = r + 8 * lh;
            lds[wib][0][m][ln] = f2bf(c[0][r]);
            lds[wib][1][m][ln] = f2bf(c[1][r]);
        }
        __syncthreads();
        Frag ap;
        ap.q[0] = *(const uint4*)&lds[wib][0][ln][lh * 8];
        ap.q[1] = *(const uint4*)&lds[wib][1][ln][lh * 8];

#pragma unroll
        for (int nt = 0; nt < 4; ++nt)
            oacc[nt] = wmma_bf16(ap, bv[nt], oacc[nt]);
    };

    Frag bkA[4], bkB[4], bvA[4], bvB[4];
    load_k(bkA, 0);
    load_v(bvA, 0);
    for (int kb = 0; kb < SS / 32; kb += 2) {
        load_k(bkB, (kb + 1) * 32);
        load_v(bvB, (kb + 1) * 32);
        attn_block(bkA, bvA, kb * 32);
        if (kb + 2 < SS / 32) {
            load_k(bkA, (kb + 2) * 32);
            load_v(bvA, (kb + 2) * 32);
        }
        attn_block(bkB, bvB, (kb + 1) * 32);
    }

    // finalize: ctx[b, s, h*64 + d] bf16
#pragma unroll
    for (int r = 0; r < 8; ++r) {
        float inv = 1.0f / rl[r];
        int m = r + 8 * lh;
        long long rowbase = ((long long)b * SS + q0 + m) * DD + h * DHH;
#pragma unroll
        for (int nt = 0; nt < 4; ++nt)
            ctx[rowbase + nt * 16 + ln] = f2bf(oacc[nt][r] * inv);
    }
}

// ---------------- output projection GEMM (+bias+residual) ----------------
// One wave per 32x64 strip; 1536 waves = 192 blocks.
__global__ __launch_bounds__(256) void k_gemm_out(
    const unsigned short* __restrict__ ctx, const unsigned short* __restrict__ wto,
    const float* __restrict__ ob, const float* __restrict__ hf,
    float* __restrict__ tmp, int layer) {
    int wave = (int)((blockIdx.x * blockDim.x + threadIdx.x) >> 5);
    int lane = threadIdx.x & 31;
    int lh = lane >> 4, ln = lane & 15;
    int mtile = wave / (DD / 64);
    int ntile = wave % (DD / 64);
    const unsigned short* W = wto + (long long)layer * DD * DD;
    const float* bias = ob + layer * DD;

    f32x8 acc[2][4];
    for (int i = 0; i < 2; ++i)
        for (int j = 0; j < 4; ++j)
            for (int r = 0; r < 8; ++r) acc[i][j][r] = 0.f;

    const unsigned short* arow0 = ctx + (long long)(mtile * 32 + ln) * DD + lh * 8;
    const unsigned short* arow1 = arow0 + 16LL * DD;
    const unsigned short* bptr[4];
#pragma unroll
    for (int nt = 0; nt < 4; ++nt)
        bptr[nt] = W + (long long)(ntile * 64 + nt * 16 + ln) * DD + lh * 8;

    auto load_a = [&](Frag* a, int k0) {
        a[0].q[0] = *(const uint4*)(arow0 + k0);
        a[0].q[1] = *(const uint4*)(arow0 + k0 + 16);
        a[1].q[0] = *(const uint4*)(arow1 + k0);
        a[1].q[1] = *(const uint4*)(arow1 + k0 + 16);
    };
    auto load_b = [&](Frag* b, int k0) {
#pragma unroll
        for (int nt = 0; nt < 4; ++nt) {
            b[nt].q[0] = *(const uint4*)(bptr[nt] + k0);
            b[nt].q[1] = *(const uint4*)(bptr[nt] + k0 + 16);
        }
    };
    auto mmas = [&](const Frag* a, const Frag* b) {
#pragma unroll
        for (int mi = 0; mi < 2; ++mi)
#pragma unroll
            for (int nt = 0; nt < 4; ++nt)
                acc[mi][nt] = wmma_bf16(a[mi], b[nt], acc[mi][nt]);
    };

    Frag aA[2], bA[4], aB[2], bB[4];
    load_a(aA, 0);
    load_b(bA, 0);
    for (int kk = 0; kk < DD / 32; kk += 2) {
        load_a(aB, (kk + 1) * 32);
        load_b(bB, (kk + 1) * 32);
        mmas(aA, bA);
        if (kk + 2 < DD / 32) {
            load_a(aA, (kk + 2) * 32);
            load_b(bA, (kk + 2) * 32);
        }
        mmas(aB, bB);
    }

#pragma unroll
    for (int mi = 0; mi < 2; ++mi)
#pragma unroll
        for (int nt = 0; nt < 4; ++nt)
#pragma unroll
            for (int r = 0; r < 8; ++r) {
                long long row = mtile * 32 + mi * 16 + r + 8 * lh;
                int n = ntile * 64 + nt * 16 + ln;
                tmp[row * DD + n] = acc[mi][nt][r] + bias[n] + hf[row * DD + n];
            }
}

// ---------------- LayerNorm ----------------
__global__ __launch_bounds__(256) void k_ln(const float* __restrict__ tmp,
                                            const float* __restrict__ g,
                                            const float* __restrict__ be,
                                            float* __restrict__ hf_out,
                                            unsigned short* __restrict__ hb_out,
                                            int layer) {
    __shared__ float red[16];
    long long row = blockIdx.x;
    const float* x = tmp + row * DD;
    int tid = threadIdx.x, w = tid >> 5, lane = tid & 31;

    float vals[3];
    float s = 0.f;
#pragma unroll
    for (int i = 0; i < 3; ++i) { vals[i] = x[tid + i * 256]; s += vals[i]; }
    for (int m = 1; m < 32; m <<= 1) s += __shfl_xor(s, m, 32);
    if (lane == 0) red[w] = s;
    __syncthreads();
    if (tid < 8) {
        float t = red[tid];
        for (int m = 1; m < 8; m <<= 1) t += __shfl_xor(t, m, 32);
        if (tid == 0) red[8] = t;
    }
    __syncthreads();
    float mean = red[8] * (1.0f / DD);
    float vs = 0.f;
#pragma unroll
    for (int i = 0; i < 3; ++i) { float d = vals[i] - mean; vs += d * d; }
    for (int m = 1; m < 32; m <<= 1) vs += __shfl_xor(vs, m, 32);
    __syncthreads();
    if (lane == 0) red[w] = vs;
    __syncthreads();
    if (tid < 8) {
        float t = red[tid];
        for (int m = 1; m < 8; m <<= 1) t += __shfl_xor(t, m, 32);
        if (tid == 0) red[9] = t;
    }
    __syncthreads();
    float rstd = rsqrtf(red[9] * (1.0f / DD) + 1e-12f);
    const float* gl = g + layer * DD;
    const float* bl = be + layer * DD;
#pragma unroll
    for (int i = 0; i < 3; ++i) {
        int d = tid + i * 256;
        float y = (vals[i] - mean) * rstd * gl[d] + bl[d];
        hf_out[row * DD + d] = y;
        hb_out[row * DD + d] = f2bf(y);
    }
}

// ---------------- host ----------------
extern "C" void kernel_launch(void* const* d_in, const int* in_sizes, int n_in,
                              void* d_out, int out_size, void* d_ws, size_t ws_size,
                              hipStream_t stream) {
    const float* hs = (const float*)d_in[0];
    const unsigned char* mask = (const unsigned char*)d_in[1];
    const int* pos = (const int*)d_in[2];
    const int* bbox = (const int*)d_in[3];
    const float* qw = (const float*)d_in[4];
    const float* qb = (const float*)d_in[5];
    const float* kw = (const float*)d_in[6];
    const float* kb = (const float*)d_in[7];
    const float* vw = (const float*)d_in[8];
    const float* vb = (const float*)d_in[9];
    const float* ow = (const float*)d_in[10];
    const float* ob = (const float*)d_in[11];
    const float* lng = (const float*)d_in[12];
    const float* lnb = (const float*)d_in[13];
    const float* rpw = (const float*)d_in[14];
    const float* rxw = (const float*)d_in[15];
    const float* ryw = (const float*)d_in[16];

    char* ws = (char*)d_ws;
    size_t off = 0;
    auto alloc = [&](size_t bytes) -> char* {
        char* p = ws + off;
        off += (bytes + 255) & ~(size_t)255;
        return p;
    };
    unsigned short* wt    = (unsigned short*)alloc(4ULL * LLAYERS * DD * DD * 2);
    float* T              = (float*)alloc((size_t)HH * 32 * 64 * 64 * 4);
    unsigned int* pidx    = (unsigned int*)alloc((size_t)BB * SS * SS * 4);
    unsigned short* hb    = (unsigned short*)alloc((size_t)BB * SS * DD * 2);
    float* hf             = (float*)alloc((size_t)BB * SS * DD * 4);
    unsigned short* qbuf  = (unsigned short*)alloc((size_t)BB * SS * DD * 2);
    unsigned short* kbuf  = (unsigned short*)alloc((size_t)BB * SS * DD * 2);
    unsigned short* vtbuf = (unsigned short*)alloc((size_t)BB * SS * DD * 2);
    unsigned short* ctx   = (unsigned short*)alloc((size_t)BB * SS * DD * 2);
    float* tmp            = (float*)alloc((size_t)BB * SS * DD * 4);

    {
        long long t1 = 4LL * LLAYERS * DD * DD;
        k_prep_w<<<(int)((t1 + 255) / 256), 256, 0, stream>>>(qw, kw, vw, ow, wt);
        int t2 = HH * 32 * 64 * 64;
        k_table<<<(t2 + 255) / 256, 256, 0, stream>>>(rpw, rxw, ryw, T);
        long long t3 = (long long)BB * SS * SS;
        k_pidx<<<(int)((t3 + 255) / 256), 256, 0, stream>>>(pos, bbox, pidx);
        int t4 = BB * SS * DD;
        k_init_h<<<(t4 + 255) / 256, 256, 0, stream>>>(hs, hf, hb);
    }
    unsigned short* wt_o = wt + 3ULL * LLAYERS * DD * DD;
    for (int l = 0; l < LLAYERS; ++l) {
        k_gemm_qkv<<<576, 256, 0, stream>>>(hb, wt, qb, kb, vb, qbuf, kbuf, vtbuf, l);
        k_attn<<<384, 256, 0, stream>>>(qbuf, kbuf, vtbuf, pidx, T, mask, ctx);
        k_gemm_out<<<192, 256, 0, stream>>>(ctx, wt_o, ob, hf, tmp, l);
        float* dst = (l == LLAYERS - 1) ? (float*)d_out : hf;
        k_ln<<<BB * SS, 256, 0, stream>>>(tmp, lng, lnb, dst, hb, l);
    }
}
